// Quantize_8881992368326
// MI455X (gfx1250) — matrix operand order repaired and checked by reading, not compiled, not run on hardware
//
#include <hip/hip_runtime.h>
#include <hip/hip_bf16.h>

typedef __attribute__((ext_vector_type(16))) _Float16 v16h;
typedef __attribute__((ext_vector_type(8)))  float    v8f;

#define DIM        64
#define NEMBED     512
#define ROWS_PER_WAVE  16
#define WAVES_PER_BLOCK 8
#define ROWS_PER_BLOCK (ROWS_PER_WAVE * WAVES_PER_BLOCK)

// LDS layout: [64 fragments][32 lanes][16 halves] f16 B-fragments (64 KB),
// then 512 f32 norms (2 KB), then 8 f32 wave partial sums.
#define SMEM_B_BYTES (DIM * NEMBED * 2)             // 65536
#define SMEM_N_OFF   SMEM_B_BYTES
#define SMEM_W_OFF   (SMEM_B_BYTES + NEMBED * 4)
#define SMEM_BYTES   (SMEM_W_OFF + WAVES_PER_BLOCK * 4)

__global__ void vq_zero_diff(float* diff) { *diff = 0.0f; }

__global__ __launch_bounds__(256) void vq_kernel(
    const float* __restrict__ x, const float* __restrict__ embed,
    float* __restrict__ out_q, float* __restrict__ out_diff,
    float* __restrict__ out_ind, int nrows)
{
  extern __shared__ __align__(16) char smem[];
  _Float16* sB = (_Float16*)smem;                 // swizzled f16 codebook fragments
  float*    sN = (float*)(smem + SMEM_N_OFF);     // ||e_j||^2 (f16-rounded operands)
  float*    sW = (float*)(smem + SMEM_W_OFF);     // per-wave diff partials

  const int tid  = threadIdx.x;
  const int lane = tid & 31;
  const int w    = tid >> 5;
  const int m    = lane & 15;   // N (column) within tile for B/C; row for A
  const int h    = lane >> 4;   // half-wave group

  const int row0 = blockIdx.x * ROWS_PER_BLOCK + w * ROWS_PER_WAVE;

  // ---- A fragments first (independent of LDS; overlaps the staging) ----
  // A layout: lane l holds row M=l%16; element i -> K = i + 16*(l/16).
  v16h a0, a1;
  {
    const float* xr = x + (size_t)(row0 + m) * DIM;
    #pragma unroll
    for (int i = 0; i < 16; ++i) {
      const int K = i + 16 * h;
      a0[i] = (_Float16)xr[K];
      a1[i] = (_Float16)xr[32 + K];
    }
  }

  // ---- Stage codebook into LDS as WMMA B-fragments (coalesced reads) ----
  // Fragment f = ct*2 + kt covers columns [ct*16, ct*16+16), K rows [kt*32, kt*32+32).
  // B layout: lane l holds column N=l%16; element i -> K = i + 16*(l/16).
  // Inverse map of linear element q = d*512 + col:
  //   kt=d/32, K=d%32, hh=K/16, i=K%16, ct=col/16, n=col%16
  //   half-offset = (ct*2+kt)*512 + (hh*16+n)*16 + i
  {
    const float4* ev = (const float4*)embed;      // 512-aligned rows; float4-safe
    for (int p = tid; p < (DIM * NEMBED) / 4; p += 256) {
      const int q   = p << 2;
      const float4 e = ev[p];
      const int d   = q >> 9;
      const int col = q & 511;                    // multiple of 4 -> same d, same ct
      const int kt = d >> 5, K = d & 31, hh = K >> 4, i = K & 15;
      const int ct = col >> 4, n = col & 15;
      _Float16* dst = sB + (ct * 2 + kt) * 512 + (hh * 16 + n) * 16 + i;
      dst[0]  = (_Float16)e.x;
      dst[16] = (_Float16)e.y;
      dst[32] = (_Float16)e.z;
      dst[48] = (_Float16)e.w;
    }
  }
  __syncthreads();

  // ---- Codebook norms from the staged f16 fragments (contiguous LDS reads) ----
  for (int j = tid; j < NEMBED; j += 256) {
    const int ct = j >> 4, n = j & 15;
    float s = 0.0f;
    #pragma unroll
    for (int kt = 0; kt < 2; ++kt)
      #pragma unroll
      for (int hh = 0; hh < 2; ++hh) {
        const _Float16* pp = sB + (ct * 2 + kt) * 512 + (hh * 16 + n) * 16;
        #pragma unroll
        for (int i = 0; i < 16; ++i) {
          const float e = (float)pp[i];
          s += e * e;
        }
      }
    sN[j] = s;
  }
  __syncthreads();

  // ---- GEMM + running argmax of  2*f.e - ||e||^2  ----
  float bestv[8];
  int   besti[8];
  #pragma unroll
  for (int r = 0; r < 8; ++r) { bestv[r] = -3.402823e38f; besti[r] = 0; }

  #pragma unroll 4
  for (int ct = 0; ct < 32; ++ct) {
    const v16h b0 = *(const v16h*)(sB + (ct * 2 + 0) * 512 + lane * 16);
    const v16h b1 = *(const v16h*)(sB + (ct * 2 + 1) * 512 + lane * 16);
    v8f c = {};
    c = __builtin_amdgcn_wmma_f32_16x16x32_f16(false, a0, false, b0, (short)0, c, false, false);
    c = __builtin_amdgcn_wmma_f32_16x16x32_f16(false, a1, false, b1, (short)0, c, false, false);
    const int j     = ct * 16 + m;     // candidate codebook column for this lane
    const float nrm = sN[j];
    #pragma unroll
    for (int r = 0; r < 8; ++r) {      // slot r -> row M = r + 8*h
      const float v = 2.0f * c[r] - nrm;
      if (v > bestv[r]) { bestv[r] = v; besti[r] = j; }
    }
  }

  // ---- Cross-lane argmax over the 16 N-lanes of each half group ----
  #pragma unroll
  for (int r = 0; r < 8; ++r) {
    float v = bestv[r]; int bi = besti[r];
    #pragma unroll
    for (int off = 1; off < 16; off <<= 1) {
      const float ov = __shfl_xor(v, off, 32);
      const int   oi = __shfl_xor(bi, off, 32);
      if (ov > v || (ov == v && oi < bi)) { v = ov; bi = oi; }  // first-match tie-break
    }
    besti[r] = bi;
  }

  // ---- Write indices (as float, per harness output buffer dtype) ----
  if (m == 0) {
    #pragma unroll
    for (int r = 0; r < 8; ++r)
      out_ind[row0 + 8 * h + r] = (float)besti[r];
  }

  // ---- Gather winning codes (exact f32, L2-resident), write quantize, MSE ----
  float lsum = 0.0f;
  #pragma unroll
  for (int t = 0; t < ROWS_PER_WAVE; ++t) {
    const int idx = __shfl(besti[t & 7], (t >> 3) * 16, 32);
    const size_t grow = (size_t)(row0 + t) * DIM;
    const float e0 = embed[(size_t)lane * NEMBED + idx];
    const float e1 = embed[(size_t)(lane + 32) * NEMBED + idx];
    out_q[grow + lane]      = e0;
    out_q[grow + lane + 32] = e1;
    const float d0 = e0 - x[grow + lane];
    const float d1 = e1 - x[grow + lane + 32];
    lsum += d0 * d0 + d1 * d1;
  }
  #pragma unroll
  for (int off = 16; off >= 1; off >>= 1)
    lsum += __shfl_xor(lsum, off, 32);
  if (lane == 0) sW[w] = lsum;
  __syncthreads();
  if (tid == 0) {
    float s = 0.0f;
    #pragma unroll
    for (int i = 0; i < WAVES_PER_BLOCK; ++i) s += sW[i];
    atomicAdd(out_diff, s / ((float)nrows * (float)DIM));
  }
}

extern "C" void kernel_launch(void* const* d_in, const int* in_sizes, int n_in,
                              void* d_out, int out_size, void* d_ws, size_t ws_size,
                              hipStream_t stream) {
  const float* x     = (const float*)d_in[0];
  const float* embed = (const float*)d_in[1];
  const int nrows = in_sizes[0] / DIM;           // 131072

  float* out      = (float*)d_out;
  float* out_q    = out;                          // [nrows*64]
  float* out_diff = out + (size_t)nrows * DIM;    // [1]
  float* out_ind  = out_diff + 1;                 // [nrows]

  vq_zero_diff<<<1, 1, 0, stream>>>(out_diff);

  const int blocks = nrows / ROWS_PER_BLOCK;      // 1024
  vq_kernel<<<blocks, 256, SMEM_BYTES, stream>>>(x, embed, out_q, out_diff, out_ind, nrows);
}
